// TopologicalLoss_11390253269372
// MI455X (gfx1250) — compile-verified, matrix operand order checked by reading, not run on hardware
//
#include <hip/hip_runtime.h>
#include <math.h>

typedef float     v2f  __attribute__((ext_vector_type(2)));
typedef float     v8f  __attribute__((ext_vector_type(8)));
typedef _Float16  v16h __attribute__((ext_vector_type(16)));

#define B_   16
#define N_   1024
#define D_   128
#define NMAT 32
#define INF_F __builtin_inff()

// workspace layout, in float units
static constexpr size_t WS_D      = 0;                               // [NMAT][N][N]  distance matrices
static constexpr size_t WS_SQ     = WS_D      + (size_t)NMAT*N_*N_;  // [NMAT][N]     squared row norms
static constexpr size_t WS_DEATHS = WS_SQ     + (size_t)NMAT*N_;     // [NMAT][1024]  H0 deaths (padded w/ +inf)
static constexpr size_t WS_MSEP   = WS_DEATHS + (size_t)NMAT*1024;   // [256]         MSE partials

// ---------------------------------------------------------------------------
// Kernel 1: squared row norms, one wave32 per row, shuffle reduce
// ---------------------------------------------------------------------------
__global__ void k_sqnorm(const float* __restrict__ x1, const float* __restrict__ x2,
                         float* __restrict__ ws) {
  int gw   = (blockIdx.x * blockDim.x + threadIdx.x) >> 5;   // global wave id
  int lane = threadIdx.x & 31;
  if (gw >= NMAT * N_) return;
  int mat = gw / N_;
  int row = gw % N_;
  const float* X  = (mat < B_) ? x1 + (size_t)mat * N_ * D_
                               : x2 + (size_t)(mat - B_) * N_ * D_;
  const float* xr = X + (size_t)row * D_;
  float s = 0.f;
  for (int k = lane; k < D_; k += 32) { float v = xr[k]; s += v * v; }
  for (int off = 16; off >= 1; off >>= 1) s += __shfl_xor(s, off, 32);
  if (lane == 0) ws[WS_SQ + (size_t)mat * N_ + row] = s;
}

// ---------------------------------------------------------------------------
// Kernel 2: Gram matrix via WMMA, then D[i][j] = sqrt(max(si+sj-2g,0)+eps),
// diagonal = +inf.  One 16x16 output tile per wave.
// ---------------------------------------------------------------------------
__global__ void k_gram_dist(const float* __restrict__ x1, const float* __restrict__ x2,
                            float* __restrict__ ws) {
  const int wib  = threadIdx.x >> 5;
  const int lane = threadIdx.x & 31;
  const long tiles_per_mat = (N_ / 16) * (N_ / 16);          // 4096
  long tile = (long)blockIdx.x * (blockDim.x >> 5) + wib;
  if (tile >= (long)NMAT * tiles_per_mat) return;

  int mat = (int)(tile / tiles_per_mat);
  int rem = (int)(tile % tiles_per_mat);
  int m0  = (rem >> 6) << 4;
  int n0  = (rem & 63) << 4;

  const float* X = (mat < B_) ? x1 + (size_t)mat * N_ * D_
                              : x2 + (size_t)(mat - B_) * N_ * D_;
  int rsel = lane & 15;
  int hi   = lane >> 4;                                      // 0 | 1 (lane half)
  const float* arow = X + (size_t)(m0 + rsel) * D_;
  const float* brow = X + (size_t)(n0 + rsel) * D_;

  v8f c = {};
#if __has_builtin(__builtin_amdgcn_wmma_f32_16x16x4_f32)
  // exact fp32 path: 16x16x4, A lane layout: VGPR0 = K{0|2}, VGPR1 = K{1|3}
  {
    int koff = hi * 2;
    for (int k0 = 0; k0 < D_; k0 += 4) {
      v2f a, b;
      a.x = arow[k0 + koff];     a.y = arow[k0 + koff + 1];
      b.x = brow[k0 + koff];     b.y = brow[k0 + koff + 1];
      c = __builtin_amdgcn_wmma_f32_16x16x4_f32(false, a, false, b,
                                                (short)0, c, false, false);
    }
  }
#else
  // fallback: f16 16x16x32 (codegen-confirmed), fp32 accumulate
  for (int k0 = 0; k0 < D_; k0 += 32) {
    v16h a, b;
    for (int v = 0; v < 8; ++v) {
      int kk = k0 + ((v & 3) * 2) + (hi ? 8 : 0) + ((v >> 2) * 16);
      a[2 * v]     = (_Float16)arow[kk];
      a[2 * v + 1] = (_Float16)arow[kk + 1];
      b[2 * v]     = (_Float16)brow[kk];
      b[2 * v + 1] = (_Float16)brow[kk + 1];
    }
    c = __builtin_amdgcn_wmma_f32_16x16x32_f16(false, a, false, b,
                                               (short)0, c, false, false);
  }
#endif

  const float* sq = ws + WS_SQ + (size_t)mat * N_;
  float*       Dm = ws + WS_D  + ((size_t)mat << 20);
  int   col = n0 + rsel;
  float sqc = sq[col];
  for (int r = 0; r < 8; ++r) {
    int   row = m0 + r + hi * 8;        // C/D layout: lanes 16-31 hold M+8
    float g   = c[r];
    float d2  = sq[row] + sqc - 2.0f * g;
    d2 = fmaxf(d2, 0.0f);
    float d = sqrtf(d2 + 1e-12f);
    if (row == col) d = INF_F;
    Dm[(size_t)row * N_ + col] = d;
  }
}

// ---------------------------------------------------------------------------
// Kernel 3: Prim's MST, one workgroup (256 threads) per matrix.
// mind[] held in registers (4 nodes/thread); argmin = shuffle + tiny LDS.
// ---------------------------------------------------------------------------
__global__ void k_prim(float* __restrict__ ws) {
  __shared__ float rv[8];
  __shared__ int   ri[8];
  __shared__ float s_bv;
  __shared__ int   s_bi;
  int mat  = blockIdx.x;
  int t    = threadIdx.x;
  int lane = t & 31, wid = t >> 5;
  const float* Dm     = ws + WS_D      + ((size_t)mat << 20);
  float*       deaths = ws + WS_DEATHS + (size_t)mat * 1024;

  float md[4]; bool tr[4];
  for (int s = 0; s < 4; ++s) {
    int j = t + 256 * s;
    md[s] = Dm[j];               // row 0 of D (D[0][0] == inf)
    tr[s] = (j == 0);
    if (j == 0) md[s] = INF_F;
  }
  if (t == 0) deaths[1023] = INF_F;   // pad: sorts last, excluded from W2 sum

  for (int step = 0; step < N_ - 1; ++step) {
    // local argmin over the 4 owned nodes (strict < keeps lowest index)
    float bv = md[0]; int bi = t;
    for (int s = 1; s < 4; ++s) {
      int j = t + 256 * s;
      if (md[s] < bv) { bv = md[s]; bi = j; }
    }
    // wave32 reduce (val, idx) with first-index tie-break
    for (int off = 16; off >= 1; off >>= 1) {
      float ov = __shfl_down(bv, off, 32);
      int   oi = __shfl_down(bi, off, 32);
      if (ov < bv || (ov == bv && oi < bi)) { bv = ov; bi = oi; }
    }
    if (lane == 0) { rv[wid] = bv; ri[wid] = bi; }
    __syncthreads();
    if (t == 0) {
      float v = rv[0]; int i = ri[0];
      for (int w = 1; w < 8; ++w)
        if (rv[w] < v || (rv[w] == v && ri[w] < i)) { v = rv[w]; i = ri[w]; }
      s_bv = v; s_bi = i;
      deaths[step] = v;          // MST edge weight == H0 death time
    }
    __syncthreads();
    int besti = s_bi;
    const float* row = Dm + (size_t)besti * N_;
    for (int s = 0; s < 4; ++s) {
      int   j  = t + 256 * s;
      float dv = row[j];         // coalesced row read (L2-resident)
      if (j == besti)      { tr[s] = true; md[s] = INF_F; }
      else if (!tr[s])     { md[s] = fminf(md[s], dv); }
    }
  }
}

// ---------------------------------------------------------------------------
// Kernel 4: bitonic sort of 1024 deaths in LDS, one workgroup per matrix
// ---------------------------------------------------------------------------
__global__ void k_sort(float* __restrict__ ws) {
  __shared__ float s[1024];
  int mat = blockIdx.x;
  int t   = threadIdx.x;
  float* deaths = ws + WS_DEATHS + (size_t)mat * 1024;
  for (int i = t; i < 1024; i += 256) s[i] = deaths[i];
  __syncthreads();
  for (int k = 2; k <= 1024; k <<= 1) {
    for (int j = k >> 1; j > 0; j >>= 1) {
      for (int i = t; i < 1024; i += 256) {
        int ixj = i ^ j;
        if (ixj > i) {
          float a = s[i], b = s[ixj];
          bool up = ((i & k) == 0);
          if ((a > b) == up) { s[i] = b; s[ixj] = a; }
        }
      }
      __syncthreads();
    }
  }
  for (int i = t; i < 1024; i += 256) deaths[i] = s[i];
}

// ---------------------------------------------------------------------------
// Kernel 5: MSE partial sums (deterministic, no atomics)
// ---------------------------------------------------------------------------
__global__ void k_mse(const float* __restrict__ x1, const float* __restrict__ x2,
                      float* __restrict__ ws) {
  __shared__ float red[256];
  int t = threadIdx.x;
  const size_t total = (size_t)B_ * N_ * D_;
  float acc = 0.f;
  for (size_t i = (size_t)blockIdx.x * 256 + t; i < total; i += (size_t)gridDim.x * 256) {
    float d = x1[i] - x2[i];
    acc += d * d;
  }
  red[t] = acc; __syncthreads();
  for (int off = 128; off >= 1; off >>= 1) {
    if (t < off) red[t] += red[t + off];
    __syncthreads();
  }
  if (t == 0) ws[WS_MSEP + blockIdx.x] = red[0];
}

// ---------------------------------------------------------------------------
// Kernel 6: finalize — MSE + 0.1 * sum_b W2(sorted d1_b, sorted d2_b)
//   (topological = (w12+w21)/2 * B = sum_b W2 since W2 is symmetric)
// ---------------------------------------------------------------------------
__global__ void k_final(const float* __restrict__ ws, float* __restrict__ out) {
  __shared__ float red[256];
  int t = threadIdx.x;
  red[t] = ws[WS_MSEP + t];
  __syncthreads();
  for (int off = 128; off >= 1; off >>= 1) {
    if (t < off) red[t] += red[t + off];
    __syncthreads();
  }
  float mse = red[0] / (float)((size_t)B_ * N_ * D_);
  __syncthreads();
  float topo = 0.f;
  for (int b = 0; b < B_; ++b) {
    const float* s1 = ws + WS_DEATHS + (size_t)b * 1024;
    const float* s2 = ws + WS_DEATHS + (size_t)(B_ + b) * 1024;
    float acc = 0.f;
    for (int i = t; i < 1023; i += 256) {   // index 1023 is the +inf pad
      float d = s1[i] - s2[i];
      acc += d * d;
    }
    red[t] = acc; __syncthreads();
    for (int off = 128; off >= 1; off >>= 1) {
      if (t < off) red[t] += red[t + off];
      __syncthreads();
    }
    if (t == 0) topo += sqrtf(red[0]);
    __syncthreads();
  }
  if (t == 0) out[0] = 1.0f * mse + 0.1f * topo;
}

// ---------------------------------------------------------------------------
extern "C" void kernel_launch(void* const* d_in, const int* in_sizes, int n_in,
                              void* d_out, int out_size, void* d_ws, size_t ws_size,
                              hipStream_t stream) {
  const float* x1 = (const float*)d_in[0];
  const float* x2 = (const float*)d_in[1];
  float* ws  = (float*)d_ws;
  float* out = (float*)d_out;

  // 1) squared norms: 32768 rows, one wave each -> 4096 blocks of 8 waves
  k_sqnorm<<<4096, 256, 0, stream>>>(x1, x2, ws);
  // 2) gram + distances: 32*4096 tiles, 8 tiles (waves) per block
  k_gram_dist<<<16384, 256, 0, stream>>>(x1, x2, ws);
  // 3) Prim MST per matrix
  k_prim<<<NMAT, 256, 0, stream>>>(ws);
  // 4) sort deaths per matrix
  k_sort<<<NMAT, 256, 0, stream>>>(ws);
  // 5) MSE partials
  k_mse<<<256, 256, 0, stream>>>(x1, x2, ws);
  // 6) combine
  k_final<<<1, 256, 0, stream>>>(ws, out);
}
